// SupervisedContrastiveLoss_23330262351971
// MI455X (gfx1250) — matrix operand order repaired and checked by reading, not compiled, not run on hardware
//
#include <hip/hip_runtime.h>
#include <hip/hip_bf16.h>
#include <stdint.h>

// ---------------- problem constants ----------------
#define NROW 4096
#define NDIM 1024
#define NNEG 5
#define TEMP 0.07f
#define CSHIFT (1.0f / 0.07f)   // fixed softmax shift; dot/T <= ~1/T so exp(d-C) <= ~1
#define EPSV 1e-12f

typedef __attribute__((ext_vector_type(16))) __bf16 v16bf;
typedef __attribute__((ext_vector_type(8)))  float  v8f;

union FragBF16 { v16bf v; uint4 q[2]; };

// ---------------- helpers ----------------
static __device__ __forceinline__ unsigned short f2bf(float x) {
    unsigned u = __float_as_uint(x);
    unsigned r = (u + 0x7FFFu + ((u >> 16) & 1u)) >> 16;   // round-to-nearest-even
    return (unsigned short)r;
}
static __device__ __forceinline__ float bf2f(unsigned short h) {
    return __uint_as_float(((unsigned)h) << 16);
}

// JAX threefry-2x32 (20 rounds)
static __device__ __forceinline__ void threefry2x32(unsigned k0, unsigned k1,
                                                    unsigned c0, unsigned c1,
                                                    unsigned& o0, unsigned& o1) {
    const unsigned R[8] = {13u, 15u, 26u, 6u, 17u, 29u, 16u, 24u};
    unsigned ks[3] = {k0, k1, k0 ^ k1 ^ 0x1BD11BDAu};
    unsigned x0 = c0 + ks[0];
    unsigned x1 = c1 + ks[1];
#pragma unroll
    for (int i = 0; i < 5; ++i) {
#pragma unroll
        for (int r = 0; r < 4; ++r) {
            x0 += x1;
            unsigned rot = R[(i * 4 + r) & 7];
            x1 = (x1 << rot) | (x1 >> (32u - rot));
            x1 ^= x0;
        }
        x0 += ks[(i + 1) % 3];
        x1 += ks[(i + 2) % 3] + (unsigned)(i + 1);
    }
    o0 = x0; o1 = x1;
}

// ---------------- kernel 1: normalize rows, emit split-bf16 (hi, lo) ----------------
__global__ __launch_bounds__(256) void normalize_split_kernel(
    const float* __restrict__ feats,
    unsigned short* __restrict__ gh,
    unsigned short* __restrict__ gl) {
    const int row = blockIdx.x;
    const int tid = threadIdx.x, lane = tid & 31, wave = tid >> 5;
    const float* x = feats + (size_t)row * NDIM;

    float ss = 0.f;
#pragma unroll
    for (int it = 0; it < NDIM / 256; ++it) {
        float v = x[tid + it * 256];
        ss += v * v;
    }
#pragma unroll
    for (int m = 1; m < 32; m <<= 1) ss += __shfl_xor(ss, m, 32);

    __shared__ float sred[8];
    __shared__ float sinv;
    if (lane == 0) sred[wave] = ss;
    __syncthreads();
    if (tid == 0) {
        float tot = 0.f;
#pragma unroll
        for (int w = 0; w < 8; ++w) tot += sred[w];
        sinv = 1.0f / fmaxf(sqrtf(tot), EPSV);
    }
    __syncthreads();
    const float inv = sinv;
#pragma unroll
    for (int it = 0; it < NDIM / 256; ++it) {
        int k = tid + it * 256;
        float y = x[k] * inv;
        unsigned short h = f2bf(y);
        float rem = y - bf2f(h);
        unsigned short l = f2bf(rem);
        gh[(size_t)row * NDIM + k] = h;
        gl[(size_t)row * NDIM + k] = l;
    }
}

// ---------------- kernel 2: threefry negative sampling (top-5 per row) ----------------
__global__ __launch_bounds__(256) void sample_negs_kernel(
    const int* __restrict__ labels,
    int* __restrict__ negidx) {
    const int row = blockIdx.x;
    const int tid = threadIdx.x;
    const unsigned HALF = 8388608u;   // (N*N)/2

    // samp_key = fold_in(key(0), 7) = threefry((0,0),(0,7))
    unsigned sk0, sk1;
    threefry2x32(0u, 0u, 0u, 7u, sk0, sk1);

    const int labi = labels[row];

    float tv[NNEG];
    int   ti[NNEG];
#pragma unroll
    for (int p = 0; p < NNEG; ++p) { tv[p] = -3.4e38f; ti[p] = 0x7FFFFFFF; }

    for (int j = tid; j < NROW; j += 256) {
        bool cand = (labels[j] != labi) || (j == row);   // mask==0 positions
        if (!cand) continue;
        unsigned e = ((unsigned)row << 12) | (unsigned)j;
        unsigned o0, o1, bits;
        if (e < HALF) { threefry2x32(sk0, sk1, e, e + HALF, o0, o1); bits = o0; }
        else          { threefry2x32(sk0, sk1, e - HALF, e, o0, o1); bits = o1; }
        float u = __uint_as_float((bits >> 9) | 0x3F800000u) - 1.0f;
        // insert (u, j) with descending value, tie-break smaller index
#pragma unroll
        for (int p = 0; p < NNEG; ++p) {
            bool better = (u > tv[p]) || (u == tv[p] && j < ti[p]);
            if (better) {
                for (int q = NNEG - 1; q > p; --q) { tv[q] = tv[q - 1]; ti[q] = ti[q - 1]; }
                tv[p] = u; ti[p] = j;
                break;
            }
        }
    }

    __shared__ float mv[256 * NNEG];
    __shared__ int   mi[256 * NNEG];
#pragma unroll
    for (int p = 0; p < NNEG; ++p) { mv[tid * NNEG + p] = tv[p]; mi[tid * NNEG + p] = ti[p]; }
    __syncthreads();

    if (tid == 0) {
        float bv[NNEG]; int bi[NNEG];
#pragma unroll
        for (int p = 0; p < NNEG; ++p) { bv[p] = -3.4e38f; bi[p] = 0x7FFFFFFF; }
        for (int c = 0; c < 256 * NNEG; ++c) {
            float v = mv[c]; int idx = mi[c];
            if (idx == 0x7FFFFFFF) continue;
#pragma unroll
            for (int p = 0; p < NNEG; ++p) {
                bool better = (v > bv[p]) || (v == bv[p] && idx < bi[p]);
                if (better) {
                    for (int q = NNEG - 1; q > p; --q) { bv[q] = bv[q - 1]; bi[q] = bi[q - 1]; }
                    bv[p] = v; bi[p] = idx;
                    break;
                }
            }
        }
#pragma unroll
        for (int p = 0; p < NNEG; ++p) negidx[row * NNEG + p] = bi[p];
    }
}

// ---------------- kernel 3: fused split-bf16 WMMA Gram + softmax statistics ----------------
// Each workgroup owns 16 rows; 8 waves split the 4096 columns, 2 col-tiles per wave
// per macro-iteration (A fragment reused across both tiles; 2 independent WMMA chains).
// stats[row*4 + {0,1,2,3}] = {rowmax(dot/T), sum_{j!=i} exp(d-C), sum mask_pos*d, count_pos}
__global__ __launch_bounds__(256) void gram_stats_kernel(
    const unsigned short* __restrict__ gh,
    const unsigned short* __restrict__ gl,
    const int* __restrict__ labels,
    float* __restrict__ stats) {
    __shared__ uint4 sAq[4096];   // 64 KB: 16 rows x (1024 hi + 1024 lo) bf16
    const int tid = threadIdx.x, lane = tid & 31, wave = tid >> 5;
    const int i0 = blockIdx.x * 16;

    // cooperative A-tile load into LDS (hi then lo), 16B chunks
#pragma unroll
    for (int it = 0; it < 16; ++it) {
        int c = tid + it * 256;
        int isLo = c >> 11;
        int cc = c & 2047;
        int row = cc >> 7, chunk = cc & 127;
        const unsigned short* src = (isLo ? gl : gh) + (size_t)(i0 + row) * NDIM;
        sAq[c] = ((const uint4*)src)[chunk];
    }
    __syncthreads();

    int labR[8];
#pragma unroll
    for (int e = 0; e < 8; ++e) labR[e] = labels[i0 + e + ((lane >= 16) ? 8 : 0)];

    float rmax[8], sexp[8], spos[8], cnt[8];
#pragma unroll
    for (int e = 0; e < 8; ++e) { rmax[e] = -3.4e38f; sexp[e] = 0.f; spos[e] = 0.f; cnt[e] = 0.f; }

    const int off_a = (lane < 16) ? 0 : 16;    // A K-half select per half-wave
    const int off_b = (lane < 16) ? 0 : 16;    // B K-half select per half-wave
    const int arow  = lane & 15;

#pragma unroll 1
    for (int t = 0; t < 16; ++t) {
        const int j0  = t * 256 + wave * 32;   // this wave's 2 tiles: [j0, j0+16), [j0+16, j0+32)
        const int gj0 = j0 + (lane & 15);
        const int gj1 = gj0 + 16;
        const int labC0 = labels[gj0];
        const int labC1 = labels[gj1];
        v8f acc0 = {0.f, 0.f, 0.f, 0.f, 0.f, 0.f, 0.f, 0.f};
        v8f acc1 = {0.f, 0.f, 0.f, 0.f, 0.f, 0.f, 0.f, 0.f};

#pragma unroll 1
        for (int seg = 0; seg < 3; ++seg) {
            // seg 0: hi*hi, seg 1: hi*lo, seg 2: lo*hi  ->  ~fp32-accurate Gram
            const int abase = (seg == 2) ? 16384 : 0;
            const unsigned short* bseg = (seg == 1) ? gl : gh;
            const unsigned short* bbase0 = bseg + (size_t)gj0 * NDIM + off_b;
            const unsigned short* bbase1 = bseg + (size_t)gj1 * NDIM + off_b;
            const int afix = abase + arow * NDIM + off_a;
#pragma unroll 2
            for (int k0 = 0; k0 < NDIM; k0 += 32) {
                FragBF16 fa, fb0, fb1;
                int ai = afix + k0;
                fa.q[0] = sAq[ai >> 3];
                fa.q[1] = sAq[(ai + 8) >> 3];
                const uint4* bp0 = (const uint4*)(bbase0 + k0);
                fb0.q[0] = bp0[0];
                fb0.q[1] = bp0[1];
                const uint4* bp1 = (const uint4*)(bbase1 + k0);
                fb1.q[0] = bp1[0];
                fb1.q[1] = bp1[1];
                acc0 = __builtin_amdgcn_wmma_f32_16x16x32_bf16(
                    false, fa.v, false, fb0.v, (short)0, acc0, false, false);
                acc1 = __builtin_amdgcn_wmma_f32_16x16x32_bf16(
                    false, fa.v, false, fb1.v, (short)0, acc1, false, false);
            }
        }

        // fused per-tile statistics (C layout: element e -> row e or e+8 by half-wave)
#pragma unroll
        for (int e = 0; e < 8; ++e) {
            int gi = i0 + e + ((lane >= 16) ? 8 : 0);
            float d0 = acc0[e] / TEMP;
            rmax[e] = fmaxf(rmax[e], d0);
            if (gi != gj0) {
                sexp[e] += expf(d0 - CSHIFT);
                if (labR[e] == labC0) { spos[e] += d0; cnt[e] += 1.f; }
            }
            float d1 = acc1[e] / TEMP;
            rmax[e] = fmaxf(rmax[e], d1);
            if (gi != gj1) {
                sexp[e] += expf(d1 - CSHIFT);
                if (labR[e] == labC1) { spos[e] += d1; cnt[e] += 1.f; }
            }
        }
    }

    // 16-lane butterfly (stays within each half-wave -> reduces over the 16 columns)
#pragma unroll
    for (int e = 0; e < 8; ++e) {
#pragma unroll
        for (int m = 1; m < 16; m <<= 1) {
            rmax[e] = fmaxf(rmax[e], __shfl_xor(rmax[e], m, 32));
            sexp[e] += __shfl_xor(sexp[e], m, 32);
            spos[e] += __shfl_xor(spos[e], m, 32);
            cnt[e]  += __shfl_xor(cnt[e], m, 32);
        }
    }

    __syncthreads();                 // done with A-tile; reuse LDS for stats
    float* sf = (float*)sAq;
    if (lane == 0 || lane == 16) {
        int rbase = (lane == 16) ? 8 : 0;
#pragma unroll
        for (int e = 0; e < 8; ++e) {
            int b = (wave * 16 + rbase + e) * 4;
            sf[b + 0] = rmax[e]; sf[b + 1] = sexp[e];
            sf[b + 2] = spos[e]; sf[b + 3] = cnt[e];
        }
    }
    __syncthreads();
    if (tid < 16) {
        float m = -3.4e38f, se = 0.f, S = 0.f, c = 0.f;
#pragma unroll
        for (int w = 0; w < 8; ++w) {
            int b = (w * 16 + tid) * 4;
            m = fmaxf(m, sf[b + 0]); se += sf[b + 1];
            S += sf[b + 2];          c += sf[b + 3];
        }
        int gi = i0 + tid;
        stats[gi * 4 + 0] = m;  stats[gi * 4 + 1] = se;
        stats[gi * 4 + 2] = S;  stats[gi * 4 + 3] = c;
    }
}

// ---------------- kernel 4: add sampled-negative contributions ----------------
__global__ __launch_bounds__(256) void neg_contrib_kernel(
    const unsigned short* __restrict__ gh,
    const unsigned short* __restrict__ gl,
    const int* __restrict__ negidx,
    float* __restrict__ stats) {
    const int row = blockIdx.x;
    const int tid = threadIdx.x, lane = tid & 31, wave = tid >> 5;
    __shared__ float sd[NNEG];

    if (wave < NNEG) {
        int j = negidx[row * NNEG + wave];
        const unsigned short* hi_i = gh + (size_t)row * NDIM;
        const unsigned short* lo_i = gl + (size_t)row * NDIM;
        const unsigned short* hi_j = gh + (size_t)j * NDIM;
        const unsigned short* lo_j = gl + (size_t)j * NDIM;
        float s = 0.f;
        for (int k = lane; k < NDIM; k += 32) {
            float ah = bf2f(hi_i[k]), al = bf2f(lo_i[k]);
            float bh = bf2f(hi_j[k]), bl = bf2f(lo_j[k]);
            s += ah * bh + ah * bl + al * bh;   // matches GEMM's 3-segment numerics
        }
#pragma unroll
        for (int m = 1; m < 32; m <<= 1) s += __shfl_xor(s, m, 32);
        if (lane == 0) sd[wave] = s / TEMP;
    }
    __syncthreads();
    if (tid == 0) {
        float S = 0.f;
#pragma unroll
        for (int k = 0; k < NNEG; ++k) S += sd[k];
        stats[row * 4 + 2] += S;           // sum mask*d
        stats[row * 4 + 3] += (float)NNEG; // mask count
    }
}

// ---------------- kernel 5: per-row loss + deterministic mean ----------------
__global__ __launch_bounds__(256) void finalize_kernel(
    const float* __restrict__ stats, float* __restrict__ out) {
    __shared__ float red[256];
    const int tid = threadIdx.x;
    float p = 0.f;
    for (int r = tid; r < NROW; r += 256) {
        float m  = stats[r * 4 + 0];
        float se = stats[r * 4 + 1];
        float S  = stats[r * 4 + 2];
        float c  = stats[r * 4 + 3];
        float denom = se * expf(CSHIFT - m);              // sum_{j!=i} exp(d - rowmax)
        float logd  = logf(denom + EPSV);
        float mlpp  = (S - c * m - c * logd) / (c + EPSV); // mean log-prob of positives
        p += -mlpp;
    }
    red[tid] = p;
    __syncthreads();
    for (int s = 128; s > 0; s >>= 1) {
        if (tid < s) red[tid] += red[tid + s];
        __syncthreads();
    }
    if (tid == 0) out[0] = red[0] / (float)NROW;
}

// ---------------- launcher ----------------
extern "C" void kernel_launch(void* const* d_in, const int* in_sizes, int n_in,
                              void* d_out, int out_size, void* d_ws, size_t ws_size,
                              hipStream_t stream) {
    (void)in_sizes; (void)n_in; (void)out_size; (void)ws_size;
    const float* feats  = (const float*)d_in[0];
    const int*   labels = (const int*)d_in[1];
    float*       out    = (float*)d_out;

    // workspace layout
    unsigned short* gh = (unsigned short*)d_ws;                       // 8 MB
    unsigned short* gl = gh + (size_t)NROW * NDIM;                    // 8 MB
    char* base = (char*)d_ws;
    int*   negidx = (int*)(base + (size_t)2 * NROW * NDIM * sizeof(unsigned short));
    float* stats  = (float*)((char*)negidx + (size_t)NROW * NNEG * sizeof(int));

    normalize_split_kernel<<<NROW, 256, 0, stream>>>(feats, gh, gl);
    sample_negs_kernel<<<NROW, 256, 0, stream>>>(labels, negidx);
    gram_stats_kernel<<<NROW / 16, 256, 0, stream>>>(gh, gl, labels, stats);
    neg_contrib_kernel<<<NROW, 256, 0, stream>>>(gh, gl, negidx, stats);
    finalize_kernel<<<1, 256, 0, stream>>>(stats, out);
}
